// DeformableVoxelTransformerEncoderLayer_45767171506681
// MI455X (gfx1250) — matrix operand order repaired and checked by reading, not compiled, not run on hardware
//
#include <hip/hip_runtime.h>
#include <hip/hip_bf16.h>

// ---------------- problem constants (structural, from reference) ----------
#define D_MODEL 64
#define HEADS   8
#define LEVELS  5
#define POINTS  4
#define DH      8
#define NQQ     20000
#define BATCH   2
#define LINTOT  45109
#define MQ      (BATCH * NQQ)       // 40000 query rows (divisible by 32)
#define VROWS   (BATCH * LINTOT)    // 90218 value rows
#define VROWS_P 90240               // padded to 32-row tiles (dead-store pad)
#define FF      1024
#define LN_EPS  1e-5f
#define WAVES   4                   // waves per block (128 threads)
#define WAVES2  2                   // K2 uses 2 waves (LDS budget)

__device__ __constant__ int   c_hs[LEVELS]  = {184, 92, 46, 23, 12};
__device__ __constant__ int   c_st[LEVELS]  = {0, 33856, 42320, 44436, 44965};
__device__ __constant__ float c_inv[LEVELS] = {1.f / 184.f, 1.f / 92.f,
                                               1.f / 46.f,  1.f / 23.f,
                                               1.f / 12.f};

typedef __attribute__((ext_vector_type(16))) __bf16 v16bf;
typedef __attribute__((ext_vector_type(8)))  __bf16 v8bf;
typedef __attribute__((ext_vector_type(8)))  float  v8f;

// ---------------- WMMA helpers (gfx1250, wave32) --------------------------
__device__ __forceinline__ v8f wmma_bf16(v16bf a, v16bf b, v8f c) {
  // D = A(16x32 bf16) * B(32x16 bf16) + C(16x16 f32)
  return __builtin_amdgcn_wmma_f32_16x16x32_bf16(false, a, false, b,
                                                 (short)0, c, false, false);
}

// A fragment (16x32 bf16) from row-major fp32 X (ld = 64).
// ISA 7.12.2 layout: lane&15 = row M, lane>>4 = K-half; per lane the 16
// elements are two contiguous 8-float runs: K = base+0..7 and base+16..23.
__device__ __forceinline__ v16bf load_A_g(const float* __restrict__ X,
                                          int rowBase, int maxRow,
                                          int k0, int lane) {
  int m = lane & 15, half = lane >> 4;
  int row = rowBase + m; if (row >= maxRow) row = maxRow - 1;
  const float4* p4 = (const float4*)(X + row * D_MODEL + k0 + half * 8);
  float4 f0 = p4[0], f1 = p4[1];   // K base + 0..7
  float4 f2 = p4[4], f3 = p4[5];   // K base + 16..23
  v16bf a;
  a[0]=(__bf16)f0.x;  a[1]=(__bf16)f0.y;  a[2]=(__bf16)f0.z;  a[3]=(__bf16)f0.w;
  a[4]=(__bf16)f1.x;  a[5]=(__bf16)f1.y;  a[6]=(__bf16)f1.z;  a[7]=(__bf16)f1.w;
  a[8]=(__bf16)f2.x;  a[9]=(__bf16)f2.y;  a[10]=(__bf16)f2.z; a[11]=(__bf16)f2.w;
  a[12]=(__bf16)f3.x; a[13]=(__bf16)f3.y; a[14]=(__bf16)f3.z; a[15]=(__bf16)f3.w;
  return a;
}

// A fragment from the sum of two fp32 tensors (q_feat + q_pos)
__device__ __forceinline__ v16bf load_A_sum(const float* __restrict__ X,
                                            const float* __restrict__ Y,
                                            int rowBase, int k0, int lane) {
  int m = lane & 15, half = lane >> 4;
  int row = rowBase + m;
  const float4* p4 = (const float4*)(X + row * D_MODEL + k0 + half * 8);
  const float4* q4 = (const float4*)(Y + row * D_MODEL + k0 + half * 8);
  float4 f0 = p4[0], f1 = p4[1], f2 = p4[4], f3 = p4[5];
  float4 g0 = q4[0], g1 = q4[1], g2 = q4[4], g3 = q4[5];
  v16bf a;
  a[0]=(__bf16)(f0.x+g0.x);  a[1]=(__bf16)(f0.y+g0.y);
  a[2]=(__bf16)(f0.z+g0.z);  a[3]=(__bf16)(f0.w+g0.w);
  a[4]=(__bf16)(f1.x+g1.x);  a[5]=(__bf16)(f1.y+g1.y);
  a[6]=(__bf16)(f1.z+g1.z);  a[7]=(__bf16)(f1.w+g1.w);
  a[8]=(__bf16)(f2.x+g2.x);  a[9]=(__bf16)(f2.y+g2.y);
  a[10]=(__bf16)(f2.z+g2.z); a[11]=(__bf16)(f2.w+g2.w);
  a[12]=(__bf16)(f3.x+g3.x); a[13]=(__bf16)(f3.y+g3.y);
  a[14]=(__bf16)(f3.z+g3.z); a[15]=(__bf16)(f3.w+g3.w);
  return a;
}

// B fragment (32x16 bf16) from a PRE-TRANSPOSED bf16 weight Wt[N][K]:
// lane&15 = col N, lane>>4 = K half; per lane K = base+0..7 and base+16..23.
__device__ __forceinline__ v16bf load_Bt(const __bf16* __restrict__ Wt,
                                         int ldk, int k0, int n0, int lane) {
  int n = lane & 15, halfk = lane >> 4;
  const v8bf* p = (const v8bf*)(Wt + (n0 + n) * ldk + k0 + 8 * halfk);
  v8bf lo = p[0], hi = p[2];
  return __builtin_shufflevector(lo, hi, 0, 1, 2, 3, 4, 5, 6, 7,
                                 8, 9, 10, 11, 12, 13, 14, 15);
}

// A fragment (16x32 bf16) from an LDS tile, row stride 32 bf16 (64 B)
__device__ __forceinline__ v16bf load_A_lds(const __bf16* base, int lane) {
  int m = lane & 15, half = lane >> 4;
  const v8bf* p = (const v8bf*)(base + m * 32 + half * 8);
  v8bf lo = p[0];   // K base + 0..7
  v8bf hi = p[2];   // K base + 16..23
  return __builtin_shufflevector(lo, hi, 0, 1, 2, 3, 4, 5, 6, 7,
                                 8, 9, 10, 11, 12, 13, 14, 15);
}

__device__ __forceinline__ void lds_wait() {
  asm volatile("s_wait_dscnt 0" ::: "memory");
}

__device__ __forceinline__ void layernorm_row(const float* v,
                                              const float* __restrict__ g,
                                              const float* __restrict__ bta,
                                              float* __restrict__ dst) {
  float m = 0.f;
#pragma unroll
  for (int i = 0; i < D_MODEL; ++i) m += v[i];
  m *= (1.f / D_MODEL);
  float var = 0.f;
#pragma unroll
  for (int i = 0; i < D_MODEL; ++i) { float d = v[i] - m; var += d * d; }
  var *= (1.f / D_MODEL);
  float inv = rsqrtf(var + LN_EPS);
#pragma unroll
  for (int i = 0; i < D_MODEL; ++i) dst[i] = (v[i] - m) * inv * g[i] + bta[i];
}

// ========== K0: transpose + convert a weight matrix to bf16 W^T[N][K] =====
__global__ __launch_bounds__(256)
void k_wt(const float* __restrict__ W, __bf16* __restrict__ Wt, int K, int N) {
  int t = blockIdx.x * 256 + threadIdx.x;
  if (t >= K * N) return;
  int k = t % K, n = t / K;
  Wt[n * K + k] = (__bf16)W[k * N + n];
}

// ================= K1: value = dense_voxel @ Wv + bv ======================
// 32 rows per wave; stores unguarded into a 32-row-padded buffer.
__global__ __launch_bounds__(32 * WAVES)
void k_value_proj(const float* __restrict__ X, const __bf16* __restrict__ WvT,
                  const float* __restrict__ bv, float* __restrict__ value) {
  int lane = threadIdx.x & 31, wave = threadIdx.x >> 5;
  int tile = blockIdx.x * WAVES + wave;
  int rowBase = tile * 32;
  if (rowBase >= VROWS) return;
  v16bf a[4];
  a[0] = load_A_g(X, rowBase,      VROWS, 0,  lane);
  a[1] = load_A_g(X, rowBase,      VROWS, 32, lane);
  a[2] = load_A_g(X, rowBase + 16, VROWS, 0,  lane);
  a[3] = load_A_g(X, rowBase + 16, VROWS, 32, lane);
  int n = lane & 15, half = lane >> 4;
#pragma unroll
  for (int nt = 0; nt < 4; ++nt) {
    int n0 = nt * 16;
    v16bf b0 = load_Bt(WvT, D_MODEL, 0,  n0, lane);
    v16bf b1 = load_Bt(WvT, D_MODEL, 32, n0, lane);
    float bias = bv[n0 + n];
#pragma unroll
    for (int mh = 0; mh < 2; ++mh) {
      v8f acc = {};
      acc = wmma_bf16(a[2 * mh],     b0, acc);
      acc = wmma_bf16(a[2 * mh + 1], b1, acc);
      int rb = rowBase + mh * 16 + 8 * half;
#pragma unroll
      for (int r = 0; r < 8; ++r)
        value[(rb + r) * D_MODEL + n0 + n] = acc[r] + bias;  // pad absorbs tail
    }
  }
}

// ===== K2: q = q_feat+q_pos; sampling locations + softmax attn weights ====
__global__ __launch_bounds__(32 * WAVES2)
void k_qproj(const float* __restrict__ qf, const float* __restrict__ qp,
             const float* __restrict__ refp,
             const __bf16* __restrict__ WoT, const float* __restrict__ bo,
             const __bf16* __restrict__ WaT, const float* __restrict__ ba,
             float* __restrict__ loc, float* __restrict__ aw) {
  __shared__ float lg[WAVES2][32][HEADS * LEVELS * POINTS];  // 32x160 logits
  int lane = threadIdx.x & 31, wave = threadIdx.x >> 5;
  int tile = blockIdx.x * WAVES2 + wave;
  int rowBase = tile * 32;
  if (rowBase >= MQ) return;                   // MQ % 32 == 0, no partials
  v16bf a[4];
  a[0] = load_A_sum(qf, qp, rowBase,      0,  lane);
  a[1] = load_A_sum(qf, qp, rowBase,      32, lane);
  a[2] = load_A_sum(qf, qp, rowBase + 16, 0,  lane);
  a[3] = load_A_sum(qf, qp, rowBase + 16, 32, lane);
  int n = lane & 15, half = lane >> 4;

  // This lane always produces coordinate c = n & 1 (n0 is even): hoist the
  // reference-point loads for its 16 rows.
  int c = n & 1;
  float refc[2][8];
#pragma unroll
  for (int mh = 0; mh < 2; ++mh)
#pragma unroll
    for (int r = 0; r < 8; ++r)
      refc[mh][r] = refp[(rowBase + mh * 16 + 8 * half + r) * 2 + c];

  // ---- offsets -> absolute sampling locations (N = 320) ----
#pragma unroll
  for (int nt = 0; nt < 20; ++nt) {
    int n0 = nt * 16;
    v16bf b0 = load_Bt(WoT, D_MODEL, 0,  n0, lane);
    v16bf b1 = load_Bt(WoT, D_MODEL, 32, n0, lane);
    int j = n0 + n;                 // ((h*5+l)*4+p)*2+c
    int l = (j % 40) >> 3;
    float inv = c_inv[l];           // normalizer (W,H); levels are square
    float bias = bo[j];
#pragma unroll
    for (int mh = 0; mh < 2; ++mh) {
      v8f acc = {};
      acc = wmma_bf16(a[2 * mh],     b0, acc);
      acc = wmma_bf16(a[2 * mh + 1], b1, acc);
      int rb = rowBase + mh * 16 + 8 * half;
#pragma unroll
      for (int r = 0; r < 8; ++r)
        loc[(rb + r) * 320 + j] = refc[mh][r] + (acc[r] + bias) * inv;
    }
  }

  // ---- attention logits (N = 160) into LDS ----
#pragma unroll
  for (int nt = 0; nt < 10; ++nt) {
    int n0 = nt * 16;
    v16bf b0 = load_Bt(WaT, D_MODEL, 0,  n0, lane);
    v16bf b1 = load_Bt(WaT, D_MODEL, 32, n0, lane);
    float bias = ba[n0 + n];
#pragma unroll
    for (int mh = 0; mh < 2; ++mh) {
      v8f acc = {};
      acc = wmma_bf16(a[2 * mh],     b0, acc);
      acc = wmma_bf16(a[2 * mh + 1], b1, acc);
#pragma unroll
      for (int r = 0; r < 8; ++r)
        lg[wave][mh * 16 + 8 * half + r][n0 + n] = acc[r] + bias;
    }
  }
  lds_wait();

  // ---- per-row softmax over 20 taps per head (all 32 lanes, 1 row each) --
  int row = rowBase + lane;
#pragma unroll
  for (int h = 0; h < HEADS; ++h) {
    const float* lr = &lg[wave][lane][h * 20];
    float mx = -3.4e38f;
#pragma unroll
    for (int t = 0; t < 20; ++t) mx = fmaxf(mx, lr[t]);
    float e[20], s = 0.f;
#pragma unroll
    for (int t = 0; t < 20; ++t) { e[t] = __expf(lr[t] - mx); s += e[t]; }
    float invs = 1.f / s;
#pragma unroll
    for (int t = 0; t < 20; ++t) aw[row * 160 + h * 20 + t] = e[t] * invs;
  }
}

// ================= K3: deformable bilinear sampling (branchless) ==========
__global__ __launch_bounds__(256)
void k_sample(const float* __restrict__ value, const float* __restrict__ loc,
              const float* __restrict__ aw, float* __restrict__ attn) {
  int t = blockIdx.x * 256 + threadIdx.x;
  if (t >= MQ * HEADS) return;
  int h = t & (HEADS - 1);
  int row = t >> 3;
  int b = row / NQQ;
  const float* lp = loc + row * 320 + h * 40;
  const float* wp = aw + row * 160 + h * 20;
  float acc[DH];
#pragma unroll
  for (int k = 0; k < DH; ++k) acc[k] = 0.f;

#pragma unroll
  for (int l = 0; l < LEVELS; ++l) {
    int Hl = c_hs[l], Wl = c_hs[l];
    const int basePix = b * LINTOT + c_st[l];
#pragma unroll
    for (int p = 0; p < POINTS; ++p) {
      float lx = lp[(l * 4 + p) * 2 + 0];
      float ly = lp[(l * 4 + p) * 2 + 1];
      float w  = wp[l * 4 + p];
      float x = lx * (float)Wl - 0.5f;
      float y = ly * (float)Hl - 0.5f;
      float x0 = floorf(x), y0 = floorf(y);
      float wx = x - x0, wy = y - y0;
      int x0i = (int)x0, y0i = (int)y0;
      // Branchless: clamp gather index, zero the corner weight when OOB
      // (matches grid_sample padding_mode='zeros').
#pragma unroll
      for (int dy = 0; dy < 2; ++dy) {
        int yi = y0i + dy;
        bool yok = (unsigned)yi < (unsigned)Hl;
        int yc = min(max(yi, 0), Hl - 1);
        float wyf = dy ? wy : 1.f - wy;
#pragma unroll
        for (int dx = 0; dx < 2; ++dx) {
          int xi = x0i + dx;
          bool ok = yok && ((unsigned)xi < (unsigned)Wl);
          int xc = min(max(xi, 0), Wl - 1);
          float cw = ok ? w * wyf * (dx ? wx : 1.f - wx) : 0.f;
          const float* vp = value + (basePix + yc * Wl + xc) * D_MODEL + h * DH;
          float4 v0 = *(const float4*)vp;
          float4 v1 = *(const float4*)(vp + 4);
          acc[0] += cw * v0.x; acc[1] += cw * v0.y;
          acc[2] += cw * v0.z; acc[3] += cw * v0.w;
          acc[4] += cw * v1.x; acc[5] += cw * v1.y;
          acc[6] += cw * v1.z; acc[7] += cw * v1.w;
        }
      }
    }
  }
  float4* op = (float4*)(attn + row * D_MODEL + h * DH);
  op[0] = make_float4(acc[0], acc[1], acc[2], acc[3]);
  op[1] = make_float4(acc[4], acc[5], acc[6], acc[7]);
}

// ========= K4: attn @ Wout + bout + q_feat residual, LayerNorm1 ===========
__global__ __launch_bounds__(32 * WAVES)
void k_outproj_ln1(const float* __restrict__ attn, const float* __restrict__ qf,
                   const __bf16* __restrict__ WoutT,
                   const float* __restrict__ bout,
                   const float* __restrict__ g1, const float* __restrict__ b1,
                   float* __restrict__ xout) {
  __shared__ float xs[WAVES][32][D_MODEL];
  int lane = threadIdx.x & 31, wave = threadIdx.x >> 5;
  int tile = blockIdx.x * WAVES + wave;
  int rowBase = tile * 32;
  if (rowBase >= MQ) return;
  v16bf a[4];
  a[0] = load_A_g(attn, rowBase,      MQ, 0,  lane);
  a[1] = load_A_g(attn, rowBase,      MQ, 32, lane);
  a[2] = load_A_g(attn, rowBase + 16, MQ, 0,  lane);
  a[3] = load_A_g(attn, rowBase + 16, MQ, 32, lane);
  int n = lane & 15, half = lane >> 4;
#pragma unroll
  for (int nt = 0; nt < 4; ++nt) {
    int n0 = nt * 16;
    v16bf b0 = load_Bt(WoutT, D_MODEL, 0,  n0, lane);
    v16bf b1 = load_Bt(WoutT, D_MODEL, 32, n0, lane);
    float bias = bout[n0 + n];
#pragma unroll
    for (int mh = 0; mh < 2; ++mh) {
      v8f acc = {};
      acc = wmma_bf16(a[2 * mh],     b0, acc);
      acc = wmma_bf16(a[2 * mh + 1], b1, acc);
      int rb = rowBase + mh * 16 + 8 * half;
#pragma unroll
      for (int r = 0; r < 8; ++r)
        xs[wave][mh * 16 + 8 * half + r][n0 + n] =
            acc[r] + bias + qf[(rb + r) * D_MODEL + n0 + n];
    }
  }
  lds_wait();
  int row = rowBase + lane;
  layernorm_row(&xs[wave][lane][0], g1, b1, xout + row * D_MODEL);
}

// ========= K5: fused FFN (x@W1 -> ReLU -> @W2) + residual + LayerNorm2 ====
__global__ __launch_bounds__(32 * WAVES)
void k_ffn_ln2(const float* __restrict__ x, const __bf16* __restrict__ W1T,
               const float* __restrict__ bff1, const __bf16* __restrict__ W2T,
               const float* __restrict__ bff2, const float* __restrict__ g2,
               const float* __restrict__ b2, float* __restrict__ out) {
  __shared__ __bf16 hb[WAVES][32][32];        // hidden tile (A-side, bf16)
  __shared__ float  ys[WAVES][32][D_MODEL];   // pre-LN result
  int lane = threadIdx.x & 31, wave = threadIdx.x >> 5;
  int tile = blockIdx.x * WAVES + wave;
  int rowBase = tile * 32;
  if (rowBase >= MQ) return;
  v16bf a[4];
  a[0] = load_A_g(x, rowBase,      MQ, 0,  lane);
  a[1] = load_A_g(x, rowBase,      MQ, 32, lane);
  a[2] = load_A_g(x, rowBase + 16, MQ, 0,  lane);
  a[3] = load_A_g(x, rowBase + 16, MQ, 32, lane);
  int n = lane & 15, half = lane >> 4;
  v8f yacc[8] = {};                            // [mh][nt] -> mh*4+nt

  for (int hc = 0; hc < FF / 32; ++hc) {       // 32 hidden chunks of 32
#pragma unroll
    for (int sub = 0; sub < 2; ++sub) {
      int n0 = hc * 32 + sub * 16;
      v16bf b0 = load_Bt(W1T, D_MODEL, 0,  n0, lane);
      v16bf b1 = load_Bt(W1T, D_MODEL, 32, n0, lane);
      float bias = bff1[n0 + n];
#pragma unroll
      for (int mh = 0; mh < 2; ++mh) {
        v8f hacc = {};
        hacc = wmma_bf16(a[2 * mh],     b0, hacc);
        hacc = wmma_bf16(a[2 * mh + 1], b1, hacc);
#pragma unroll
        for (int r = 0; r < 8; ++r)
          hb[wave][mh * 16 + 8 * half + r][sub * 16 + n] =
              (__bf16)fmaxf(hacc[r] + bias, 0.f);
      }
    }
    lds_wait();                                // C-layout -> A-layout via LDS
    v16bf ah0 = load_A_lds(&hb[wave][0][0],  lane);
    v16bf ah1 = load_A_lds(&hb[wave][16][0], lane);
    int k0 = hc * 32;
#pragma unroll
    for (int nt = 0; nt < 4; ++nt) {
      v16bf bw = load_Bt(W2T, FF, k0, nt * 16, lane);
      yacc[nt]     = wmma_bf16(ah0, bw, yacc[nt]);
      yacc[4 + nt] = wmma_bf16(ah1, bw, yacc[4 + nt]);
    }
  }

#pragma unroll
  for (int nt = 0; nt < 4; ++nt) {
    int n0 = nt * 16;
    float bias = bff2[n0 + n];
#pragma unroll
    for (int mh = 0; mh < 2; ++mh) {
      int rb = rowBase + mh * 16 + 8 * half;
#pragma unroll
      for (int r = 0; r < 8; ++r)
        ys[wave][mh * 16 + 8 * half + r][n0 + n] =
            yacc[mh * 4 + nt][r] + bias + x[(rb + r) * D_MODEL + n0 + n];
    }
  }
  lds_wait();
  int row = rowBase + lane;
  layernorm_row(&ys[wave][lane][0], g2, b2, out + row * D_MODEL);
}

// ======================== host-side launcher ==============================
extern "C" void kernel_launch(void* const* d_in, const int* in_sizes, int n_in,
                              void* d_out, int out_size, void* d_ws,
                              size_t ws_size, hipStream_t stream) {
  const float* q_feat = (const float*)d_in[0];
  const float* dvf    = (const float*)d_in[1];
  const float* refp   = (const float*)d_in[2];
  const float* q_pos  = (const float*)d_in[3];
  // d_in[4]/d_in[5]: spatial_shapes / level_start_index (structural; baked in)
  const float* Wv   = (const float*)d_in[6];
  const float* bv   = (const float*)d_in[7];
  const float* Wo   = (const float*)d_in[8];
  const float* bo   = (const float*)d_in[9];
  const float* Wa   = (const float*)d_in[10];
  const float* ba   = (const float*)d_in[11];
  const float* Wout = (const float*)d_in[12];
  const float* bout = (const float*)d_in[13];
  const float* g1   = (const float*)d_in[14];
  const float* b1   = (const float*)d_in[15];
  const float* W1   = (const float*)d_in[16];
  const float* bff1 = (const float*)d_in[17];
  const float* W2   = (const float*)d_in[18];
  const float* bff2 = (const float*)d_in[19];
  const float* g2   = (const float*)d_in[20];
  const float* b2   = (const float*)d_in[21];

  char* ws = (char*)d_ws;
  size_t off = 0;
  auto alloc = [&](size_t bytes) -> void* {
    void* p = ws + off;
    off = (off + bytes + 255) & ~(size_t)255;
    return p;
  };
  float*  value = (float*)alloc((size_t)VROWS_P * D_MODEL * sizeof(float));
  float*  loc   = (float*)alloc((size_t)MQ * 320 * sizeof(float));
  float*  aw    = (float*)alloc((size_t)MQ * 160 * sizeof(float));
  float*  attn  = (float*)alloc((size_t)MQ * D_MODEL * sizeof(float));
  float*  xbuf  = (float*)alloc((size_t)MQ * D_MODEL * sizeof(float));
  __bf16* WvT   = (__bf16*)alloc((size_t)64 * 64 * sizeof(__bf16));
  __bf16* WoT   = (__bf16*)alloc((size_t)320 * 64 * sizeof(__bf16));
  __bf16* WaT   = (__bf16*)alloc((size_t)160 * 64 * sizeof(__bf16));
  __bf16* WoutT = (__bf16*)alloc((size_t)64 * 64 * sizeof(__bf16));
  __bf16* W1T   = (__bf16*)alloc((size_t)1024 * 64 * sizeof(__bf16));
  __bf16* W2T   = (__bf16*)alloc((size_t)64 * 1024 * sizeof(__bf16));
  (void)ws_size; (void)in_sizes; (void)n_in; (void)out_size;

  // -- K0: weight transpose/convert passes (bf16 W^T, contiguous-K lanes) --
  k_wt<<<(64 * 64 + 255) / 256, 256, 0, stream>>>(Wv, WvT, 64, 64);
  k_wt<<<(64 * 320 + 255) / 256, 256, 0, stream>>>(Wo, WoT, 64, 320);
  k_wt<<<(64 * 160 + 255) / 256, 256, 0, stream>>>(Wa, WaT, 64, 160);
  k_wt<<<(64 * 64 + 255) / 256, 256, 0, stream>>>(Wout, WoutT, 64, 64);
  k_wt<<<(64 * 1024 + 255) / 256, 256, 0, stream>>>(W1, W1T, 64, 1024);
  k_wt<<<(1024 * 64 + 255) / 256, 256, 0, stream>>>(W2, W2T, 1024, 64);

  int vtiles = (VROWS + 31) / 32;   // 2820
  int qtiles = MQ / 32;             // 1250

  k_value_proj<<<(vtiles + WAVES - 1) / WAVES, 32 * WAVES, 0, stream>>>(
      dvf, WvT, bv, value);
  k_qproj<<<(qtiles + WAVES2 - 1) / WAVES2, 32 * WAVES2, 0, stream>>>(
      q_feat, q_pos, refp, WoT, bo, WaT, ba, loc, aw);
  int sthreads = MQ * HEADS;
  k_sample<<<(sthreads + 255) / 256, 256, 0, stream>>>(value, loc, aw, attn);
  k_outproj_ln1<<<(qtiles + WAVES - 1) / WAVES, 32 * WAVES, 0, stream>>>(
      attn, q_feat, WoutT, bout, g1, b1, xbuf);
  k_ffn_ln2<<<(qtiles + WAVES - 1) / WAVES, 32 * WAVES, 0, stream>>>(
      xbuf, W1T, bff1, W2T, bff2, g2, b2, (float*)d_out);
}
